// SMILESGenerator_22333829939908
// MI455X (gfx1250) — matrix operand order repaired and checked by reading, not compiled, not run on hardware
//
#include <hip/hip_runtime.h>
#include <hip/hip_bf16.h>

#define BB      256
#define TT      128
#define HH      1024
#define VIN     512
#define NVOCAB  64
#define NHEADS  4
#define H3      3072
#define DH      256   // head dim

typedef __attribute__((ext_vector_type(16))) __bf16 v16bf;
typedef __attribute__((ext_vector_type(8)))  __bf16 v8bf;
typedef __attribute__((ext_vector_type(8)))  float  v8f;

#define WMMA_BF16(a, b, c) \
  __builtin_amdgcn_wmma_f32_16x16x32_bf16(false, (a), false, (b), (short)0, (c), false, false)

// ---------------------------------------------------------------------------
// CDNA5 wave32 WMMA bf16 operand layout as used here (16x16x32; contiguous k is this code's choice, not AMD's table order):
//   A: lane l -> row m=l&15; per 32-K step, elements 0..15 = K[(l>>4)*16 .. +16) (contiguous)
//   B: lane l -> row n=l&15; elements 0..15 = K[(l>>4)*16 .. +16) (contiguous)
//   C: lane l -> col n=l&15; vgpr i -> row m = i + 8*(l>>4)
//
// Both tile helpers are explicitly software-pipelined: operands for k-step
// i+1 are loaded before the WMMA consuming k-step i, creating a true
// double-buffer dependence so waits become partial instead of full drains.
// ---------------------------------------------------------------------------
__device__ __forceinline__ v8f wmma_tile(const __bf16* A, int lda,
                                         const __bf16* B, int ldb,
                                         int K, v8f acc) {
  const int l  = threadIdx.x & 31;
  const int r  = l & 15;
  const int hi = l >> 4;
  const __bf16* ap = A + (size_t)r * lda + hi * 16;
  const __bf16* bp = B + (size_t)r * ldb + hi * 16;
  v16bf a, b;
  a = *(const v16bf*)(ap);
  b = *(const v16bf*)(bp);
#pragma unroll 4
  for (int k0 = 32; k0 < K; k0 += 32) {
    v16bf an, bn;
    an = *(const v16bf*)(ap + k0);
    bn = *(const v16bf*)(bp + k0);
    acc = WMMA_BF16(a, b, acc);
    a = an; b = bn;
  }
  acc = WMMA_BF16(a, b, acc);
  return acc;
}

// 16x64 output tile per wave: one A fetch feeds 4 WMMAs per k-step.
__device__ __forceinline__ void wmma_tile4(const __bf16* __restrict__ A, int lda,
                                           const __bf16* __restrict__ B, int ldb,
                                           int K, v8f acc[4]) {
  const int l  = threadIdx.x & 31;
  const int r  = l & 15;
  const int hi = l >> 4;
  const __bf16* ap = A + (size_t)r * lda + hi * 16;
  const __bf16* bp = B + (size_t)r * ldb + hi * 16;
  v16bf a, b0, b1, b2, b3;
  a = *(const v16bf*)(ap);
  b0 = *(const v16bf*)(bp);
  b1 = *(const v16bf*)(bp + (size_t)16 * ldb);
  b2 = *(const v16bf*)(bp + (size_t)32 * ldb);
  b3 = *(const v16bf*)(bp + (size_t)48 * ldb);
#pragma unroll 2
  for (int k0 = 32; k0 < K; k0 += 32) {
    v16bf an, b0n, b1n, b2n, b3n;
    an = *(const v16bf*)(ap + k0);
    b0n = *(const v16bf*)(bp + k0);
    b1n = *(const v16bf*)(bp + (size_t)16 * ldb + k0);
    b2n = *(const v16bf*)(bp + (size_t)32 * ldb + k0);
    b3n = *(const v16bf*)(bp + (size_t)48 * ldb + k0);
    acc[0] = WMMA_BF16(a, b0, acc[0]);
    acc[1] = WMMA_BF16(a, b1, acc[1]);
    acc[2] = WMMA_BF16(a, b2, acc[2]);
    acc[3] = WMMA_BF16(a, b3, acc[3]);
    a = an; b0 = b0n; b1 = b1n; b2 = b2n; b3 = b3n;
  }
  acc[0] = WMMA_BF16(a, b0, acc[0]);
  acc[1] = WMMA_BF16(a, b1, acc[1]);
  acc[2] = WMMA_BF16(a, b2, acc[2]);
  acc[3] = WMMA_BF16(a, b3, acc[3]);
}

// gfx1250 async global->LDS copy, 16B per lane, tracked by ASYNCcnt.
__device__ __forceinline__ void async_copy_b128(unsigned lds_off, const void* gaddr) {
  asm volatile("global_load_async_to_lds_b128 %0, %1, off"
               :: "v"(lds_off), "v"(gaddr) : "memory");
}
__device__ __forceinline__ void wait_async0() {
  asm volatile("s_wait_asynccnt 0x0" ::: "memory");
}

// --------------------------- small helper kernels ---------------------------
__global__ void k_cvt(const float* __restrict__ s, __bf16* __restrict__ d, size_t n) {
  size_t i = (size_t)blockIdx.x * blockDim.x + threadIdx.x;
  if (i < n) d[i] = (__bf16)s[i];
}

__global__ void k_embed(const int* __restrict__ tgt, const float* __restrict__ emb,
                        __bf16* __restrict__ xb) {
  const int t = blockIdx.x, b = blockIdx.y;
  const int tok = (t == 0) ? 1 : tgt[b * TT + (t - 1)];
  const float* e = emb + (size_t)tok * HH;
  __bf16* xo = xb + ((size_t)b * TT + t) * HH;
  for (int k = threadIdx.x; k < HH; k += blockDim.x) xo[k] = (__bf16)e[k];
}

// h0 = vector @ W_h^T + b_h ; broadcast to both layers (fp32 state + bf16 buf0)
__global__ void k_h0(const __bf16* __restrict__ vecb, const __bf16* __restrict__ Whb,
                     const float* __restrict__ b_h,
                     float* __restrict__ h0f, float* __restrict__ h1f,
                     __bf16* __restrict__ hb0, __bf16* __restrict__ hb1) {
  const int w  = threadIdx.x >> 5;
  const int n0 = (blockIdx.x * 4 + w) * 64;
  const int m0 = blockIdx.y * 16;
  v8f acc[4] = {};
  wmma_tile4(vecb + (size_t)m0 * VIN, VIN, Whb + (size_t)n0 * VIN, VIN, VIN, acc);
  const int l = threadIdx.x & 31, col = l & 15, rb = (l >> 4) * 8;
#pragma unroll
  for (int j = 0; j < 4; ++j)
#pragma unroll
    for (int i = 0; i < 8; ++i) {
      const int m = m0 + rb + i, n = n0 + j * 16 + col;
      float v = acc[j][i] + b_h[n];
      h0f[(size_t)m * HH + n] = v;
      h1f[(size_t)m * HH + n] = v;
      __bf16 bv = (__bf16)v;
      hb0[(size_t)m * HH + n] = bv;
      hb1[(size_t)m * HH + n] = bv;
    }
}

// One GRU layer, one timestep. A matrices staged block-wide in LDS via the
// gfx1250 async-to-LDS path; each wave computes a 16x16 h-tile = 6 WMMA GEMMs.
__global__ void k_gru_step(const __bf16* __restrict__ Ain, int lda,
                           const __bf16* __restrict__ hprev,
                           const __bf16* __restrict__ Wihl,
                           const __bf16* __restrict__ Whhl,
                           const float* __restrict__ bihl,
                           const float* __restrict__ bhhl,
                           float* __restrict__ hstate,
                           __bf16* __restrict__ hnext,
                           float* __restrict__ outt,     // nullptr for layer 0
                           __bf16* __restrict__ outbt) {
  __shared__ __bf16 As[16 * HH];   // 32 KB: input rows
  __shared__ __bf16 Hs[16 * HH];   // 32 KB: hidden rows
  const int m0 = blockIdx.y * 16;

  // Stage 16 rows x 2KB of both A matrices into LDS (16B per lane per issue).
  {
    const char* ga = (const char*)(Ain + (size_t)m0 * lda);
    const char* gh = (const char*)(hprev + (size_t)m0 * HH);
    const unsigned la = (unsigned)(uintptr_t)&As[0];
    const unsigned lh = (unsigned)(uintptr_t)&Hs[0];
    for (int v = threadIdx.x; v < 2048; v += 128) {
      const int row = v >> 7;            // 128 x 16B per row
      const int off = (v & 127) << 4;
      async_copy_b128(la + (unsigned)(row * 2048 + off),
                      ga + (size_t)row * lda * 2 + off);
      async_copy_b128(lh + (unsigned)(row * 2048 + off),
                      gh + (size_t)row * HH * 2 + off);
    }
    wait_async0();
    __syncthreads();
  }

  const int w  = threadIdx.x >> 5;
  const int n0 = (blockIdx.x * 4 + w) * 16;
  v8f gi0 = {}, gi1 = {}, gi2 = {}, gh0 = {}, gh1 = {}, gh2 = {};
  gi0 = wmma_tile(As, HH, Wihl + (size_t)(0 * HH + n0) * HH, HH, HH, gi0);
  gi1 = wmma_tile(As, HH, Wihl + (size_t)(1 * HH + n0) * HH, HH, HH, gi1);
  gi2 = wmma_tile(As, HH, Wihl + (size_t)(2 * HH + n0) * HH, HH, HH, gi2);
  gh0 = wmma_tile(Hs, HH, Whhl + (size_t)(0 * HH + n0) * HH, HH, HH, gh0);
  gh1 = wmma_tile(Hs, HH, Whhl + (size_t)(1 * HH + n0) * HH, HH, HH, gh1);
  gh2 = wmma_tile(Hs, HH, Whhl + (size_t)(2 * HH + n0) * HH, HH, HH, gh2);

  const int l = threadIdx.x & 31, col = l & 15, rb = (l >> 4) * 8;
#pragma unroll
  for (int i = 0; i < 8; ++i) {
    const int m = m0 + rb + i, n = n0 + col;
    float ir  = gi0[i] + bihl[n];
    float iz  = gi1[i] + bihl[HH + n];
    float inn = gi2[i] + bihl[2 * HH + n];
    float hr  = gh0[i] + bhhl[n];
    float hz  = gh1[i] + bhhl[HH + n];
    float hn  = gh2[i] + bhhl[2 * HH + n];
    float r  = 1.f / (1.f + __expf(-(ir + hr)));
    float z  = 1.f / (1.f + __expf(-(iz + hz)));
    float nn = tanhf(inn + r * hn);
    float hold = hstate[(size_t)m * HH + n];
    float hv = (1.f - z) * nn + z * hold;
    hstate[(size_t)m * HH + n] = hv;
    hnext[(size_t)m * HH + n]  = (__bf16)hv;
    if (outt) {
      outt[(size_t)m * TT * HH + n]  = hv;
      outbt[(size_t)m * TT * HH + n] = (__bf16)hv;
    }
  }
}

// qkv = out @ Wqkv^T + bqkv ; q,k kept [B,T,H]; v scattered to [B*HEADS, d, T]
__global__ void k_qkv(const __bf16* __restrict__ outb, const __bf16* __restrict__ Wqkvb,
                      const float* __restrict__ bqkv,
                      __bf16* __restrict__ qb, __bf16* __restrict__ kb,
                      __bf16* __restrict__ vtb) {
  const int w  = threadIdx.x >> 5;
  const int n0 = (blockIdx.x * 4 + w) * 64;   // 0..3071
  const int m0 = blockIdx.y * 16;             // 0..32767
  v8f acc[4] = {};
  wmma_tile4(outb + (size_t)m0 * HH, HH, Wqkvb + (size_t)n0 * HH, HH, HH, acc);
  const int l = threadIdx.x & 31, col = l & 15, rb = (l >> 4) * 8;
#pragma unroll
  for (int j = 0; j < 4; ++j)
#pragma unroll
    for (int i = 0; i < 8; ++i) {
      const int m = m0 + rb + i, n = n0 + j * 16 + col;
      __bf16 bv = (__bf16)(acc[j][i] + bqkv[n]);
      if (n < HH) {
        qb[(size_t)m * HH + n] = bv;
      } else if (n < 2 * HH) {
        kb[(size_t)m * HH + (n - HH)] = bv;
      } else {
        const int jj = n - 2 * HH, head = jj >> 8, jd = jj & 255;
        const int b = m / TT, t = m % TT;
        vtb[((size_t)(b * NHEADS + head) * DH + jd) * TT + t] = bv;
      }
    }
}

// Per (b,head): scores = q k^T / 16, causal softmax, store bf16 probabilities.
__global__ void k_attn(const __bf16* __restrict__ qb, const __bf16* __restrict__ kb,
                       __bf16* __restrict__ attnb) {
  __shared__ float s[TT * TT];                 // 64 KB
  const int bh = blockIdx.x, b = bh / NHEADS, head = bh % NHEADS;
  const __bf16* qbase = qb + (size_t)b * TT * HH + head * DH;
  const __bf16* kbase = kb + (size_t)b * TT * HH + head * DH;
  const int w = threadIdx.x >> 5;              // 8 waves -> 8 M-tiles
  const int m0 = w * 16;
  const int l = threadIdx.x & 31, col = l & 15, rb = (l >> 4) * 8;
  for (int nt = 0; nt < 8; ++nt) {
    const int n0 = nt * 16;
    v8f acc = {};
    acc = wmma_tile(qbase + (size_t)m0 * HH, HH, kbase + (size_t)n0 * HH, HH, DH, acc);
#pragma unroll
    for (int i = 0; i < 8; ++i)
      s[(m0 + rb + i) * TT + n0 + col] = acc[i] * 0.0625f;  // 1/sqrt(256)
  }
  __syncthreads();
  const int row = threadIdx.x;
  if (row < TT) {
    float mx = -3.0e38f;
    for (int n = 0; n <= row; ++n) mx = fmaxf(mx, s[row * TT + n]);
    float sum = 0.f;
    for (int n = 0; n <= row; ++n) {
      float e = __expf(s[row * TT + n] - mx);
      s[row * TT + n] = e;
      sum += e;
    }
    const float inv = 1.f / sum;
    __bf16* arow = attnb + ((size_t)bh * TT + row) * TT;
    for (int n = 0; n < TT; ++n)
      arow[n] = (__bf16)(n <= row ? s[row * TT + n] * inv : 0.f);
  }
}

// ctx = attn @ v  (K = T = 128), output [B,T,H] bf16
__global__ void k_ctx(const __bf16* __restrict__ attnb, const __bf16* __restrict__ vtb,
                      __bf16* __restrict__ ctxb) {
  const int bh = blockIdx.y, b = bh / NHEADS, head = bh % NHEADS;
  const int w  = threadIdx.x >> 5;
  const int n0 = w * 64;                      // 4 waves cover d=256
  const int m0 = blockIdx.x * 16;             // 0..127
  const __bf16* A  = attnb + (size_t)bh * TT * TT;
  const __bf16* Bm = vtb + (size_t)bh * DH * TT;
  v8f acc[4] = {};
  wmma_tile4(A + (size_t)m0 * TT, TT, Bm + (size_t)n0 * TT, TT, TT, acc);
  const int l = threadIdx.x & 31, col = l & 15, rb = (l >> 4) * 8;
#pragma unroll
  for (int j = 0; j < 4; ++j)
#pragma unroll
    for (int i = 0; i < 8; ++i) {
      const int m = m0 + rb + i, n = n0 + j * 16 + col;
      ctxb[((size_t)b * TT + m) * HH + head * DH + n] = (__bf16)acc[j][i];
    }
}

// combined = out + ctx @ Wo^T + bo  -> bf16
__global__ void k_wo(const __bf16* __restrict__ ctxb, const __bf16* __restrict__ Wob,
                     const float* __restrict__ bo, const float* __restrict__ outf,
                     __bf16* __restrict__ combinedb) {
  const int w  = threadIdx.x >> 5;
  const int n0 = (blockIdx.x * 4 + w) * 64;   // 0..1023
  const int m0 = blockIdx.y * 16;             // 0..32767
  v8f acc[4] = {};
  wmma_tile4(ctxb + (size_t)m0 * HH, HH, Wob + (size_t)n0 * HH, HH, HH, acc);
  const int l = threadIdx.x & 31, col = l & 15, rb = (l >> 4) * 8;
#pragma unroll
  for (int j = 0; j < 4; ++j)
#pragma unroll
    for (int i = 0; i < 8; ++i) {
      const int m = m0 + rb + i, n = n0 + j * 16 + col;
      float v = acc[j][i] + bo[n] + outf[(size_t)m * HH + n];
      combinedb[(size_t)m * HH + n] = (__bf16)v;
    }
}

// logits = combined @ Wout^T + bout -> fp32 d_out [B*T, 64]
__global__ void k_logits(const __bf16* __restrict__ combinedb,
                         const __bf16* __restrict__ Woutb,
                         const float* __restrict__ bout, float* __restrict__ outp) {
  const int m0 = blockIdx.x * 16;
  v8f acc[4] = {};
  wmma_tile4(combinedb + (size_t)m0 * HH, HH, Woutb, HH, HH, acc);
  const int l = threadIdx.x & 31, col = l & 15, rb = (l >> 4) * 8;
#pragma unroll
  for (int j = 0; j < 4; ++j)
#pragma unroll
    for (int i = 0; i < 8; ++i) {
      const int m = m0 + rb + i, n = j * 16 + col;
      outp[(size_t)m * NVOCAB + n] = acc[j][i] + bout[n];
    }
}

// ---------------------------------------------------------------------------
extern "C" void kernel_launch(void* const* d_in, const int* in_sizes, int n_in,
                              void* d_out, int out_size, void* d_ws, size_t ws_size,
                              hipStream_t stream) {
  const float* vector = (const float*)d_in[0];
  const int*   tgt    = (const int*)d_in[1];
  const float* W_h    = (const float*)d_in[2];
  const float* b_h    = (const float*)d_in[3];
  const float* emb    = (const float*)d_in[4];
  const float* Wih    = (const float*)d_in[5];
  const float* Whh    = (const float*)d_in[6];
  const float* bih    = (const float*)d_in[7];
  const float* bhh    = (const float*)d_in[8];
  const float* Wqkv   = (const float*)d_in[9];
  const float* bqkv   = (const float*)d_in[10];
  const float* Wo     = (const float*)d_in[11];
  const float* bo     = (const float*)d_in[12];
  const float* Wout   = (const float*)d_in[13];
  const float* bout   = (const float*)d_in[14];
  float* logits = (float*)d_out;
  (void)in_sizes; (void)n_in; (void)out_size; (void)ws_size;

  // ---- workspace carving (256B aligned sub-buffers) ----
  char* p = (char*)d_ws;
  auto carve = [&](size_t elems, size_t elsz) -> void* {
    void* r = (void*)p;
    p += (elems * elsz + 255) & ~(size_t)255;
    return r;
  };
  const size_t BTH = (size_t)BB * TT * HH;
  __bf16* vecb  = (__bf16*)carve((size_t)BB * VIN, 2);
  __bf16* Whb   = (__bf16*)carve((size_t)HH * VIN, 2);
  __bf16* Wihb  = (__bf16*)carve((size_t)2 * H3 * HH, 2);
  __bf16* Whhb  = (__bf16*)carve((size_t)2 * H3 * HH, 2);
  __bf16* Wqkvb = (__bf16*)carve((size_t)H3 * HH, 2);
  __bf16* Wob   = (__bf16*)carve((size_t)HH * HH, 2);
  __bf16* Woutb = (__bf16*)carve((size_t)NVOCAB * HH, 2);
  __bf16* xb    = (__bf16*)carve(BTH, 2);                 // reused as ctxb later
  __bf16* hb0a  = (__bf16*)carve((size_t)BB * HH, 2);     // layer0 ping-pong
  __bf16* hb0b  = (__bf16*)carve((size_t)BB * HH, 2);
  __bf16* hb1a  = (__bf16*)carve((size_t)BB * HH, 2);     // layer1 ping-pong
  __bf16* hb1b  = (__bf16*)carve((size_t)BB * HH, 2);
  float*  h0f   = (float*)carve((size_t)BB * HH, 4);
  float*  h1f   = (float*)carve((size_t)BB * HH, 4);
  float*  outf  = (float*)carve(BTH, 4);
  __bf16* outb  = (__bf16*)carve(BTH, 2);
  __bf16* qb    = (__bf16*)carve(BTH, 2);                 // reused as combinedb
  __bf16* kb    = (__bf16*)carve(BTH, 2);
  __bf16* vtb   = (__bf16*)carve(BTH, 2);
  __bf16* attnb = (__bf16*)carve((size_t)BB * NHEADS * TT * TT, 2);
  __bf16* ctxb      = xb;                                 // reuse (x dead after GRU)
  __bf16* combinedb = qb;                                 // reuse (q dead after attn)

  // ---- weight / input conversions to bf16 ----
  auto conv = [&](const float* s, __bf16* d, size_t n) {
    unsigned blocks = (unsigned)((n + 255) / 256);
    k_cvt<<<dim3(blocks), dim3(256), 0, stream>>>(s, d, n);
  };
  conv(vector, vecb, (size_t)BB * VIN);
  conv(W_h,   Whb,   (size_t)HH * VIN);
  conv(Wih,   Wihb,  (size_t)2 * H3 * HH);
  conv(Whh,   Whhb,  (size_t)2 * H3 * HH);
  conv(Wqkv,  Wqkvb, (size_t)H3 * HH);
  conv(Wo,    Wob,   (size_t)HH * HH);
  conv(Wout,  Woutb, (size_t)NVOCAB * HH);

  // ---- embedding (shifted tokens, START=1) ----
  k_embed<<<dim3(TT, BB), dim3(256), 0, stream>>>(tgt, emb, xb);

  // ---- h0 GEMM ----
  k_h0<<<dim3(4, 16), dim3(128), 0, stream>>>(vecb, Whb, b_h, h0f, h1f, hb0a, hb1a);

  // ---- GRU scan: 128 steps x 2 layers ----
  __bf16* hb0[2] = {hb0a, hb0b};
  __bf16* hb1[2] = {hb1a, hb1b};
  const dim3 ggrid(16, 16), gblk(128);
  for (int t = 0; t < TT; ++t) {
    const int cur = t & 1, nxt = cur ^ 1;
    // layer 0: input = x[:, t, :]
    k_gru_step<<<ggrid, gblk, 0, stream>>>(
        xb + (size_t)t * HH, TT * HH, hb0[cur],
        Wihb, Whhb, bih, bhh, h0f, hb0[nxt], nullptr, nullptr);
    // layer 1: input = layer0's fresh h
    k_gru_step<<<ggrid, gblk, 0, stream>>>(
        hb0[nxt], HH, hb1[cur],
        Wihb + (size_t)H3 * HH, Whhb + (size_t)H3 * HH,
        bih + H3, bhh + H3, h1f, hb1[nxt],
        outf + (size_t)t * HH, outb + (size_t)t * HH);
  }

  // ---- QKV projection ----
  k_qkv<<<dim3(12, 2048), dim3(128), 0, stream>>>(outb, Wqkvb, bqkv, qb, kb, vtb);

  // ---- causal attention scores + softmax per (b, head) ----
  k_attn<<<dim3(BB * NHEADS), dim3(256), 0, stream>>>(qb, kb, attnb);

  // ---- context GEMM ----
  k_ctx<<<dim3(8, BB * NHEADS), dim3(128), 0, stream>>>(attnb, vtb, ctxb);

  // ---- Wo projection + residual add ----
  k_wo<<<dim3(4, 2048), dim3(128), 0, stream>>>(ctxb, Wob, bo, outf, combinedb);

  // ---- output logits ----
  k_logits<<<dim3(2048), dim3(32), 0, stream>>>(combinedb, Woutb, bout, logits);
}
